// ContrastiveMultimodalAlignmentLoss_68856915690211
// MI455X (gfx1250) — matrix operand order repaired and checked by reading, not compiled, not run on hardware
//
#include <hip/hip_runtime.h>

// ---------------------------------------------------------------------------
// ContrastiveMultimodalAlignmentLoss for MI455X (gfx1250, wave32, WMMA).
// B=128, T=32, V=64, D=512.  Core contraction: v_wmma_f32_16x16x32_bf16.
// Each wave owns a 16x32 output strip (2 accumulators share one A fragment)
// -> 3 global_load_b128 per WMMA instead of 4.  Butterfly reductions use
// ds_swizzle_b32 XOR patterns (single DS op, no index VALU).
// ---------------------------------------------------------------------------

typedef __attribute__((ext_vector_type(16))) __bf16 v16bf;
typedef __attribute__((ext_vector_type(8)))  __bf16 v8bf;
typedef __attribute__((ext_vector_type(8)))  float  v8f;

#define NB 128
#define TT 32
#define VV 64
#define DD 512

// ds_swizzle group-of-32 XOR butterfly: offset = (xor_mask<<10) | and_mask(0x1F)
#define SWZ_MAXF(x, imm) \
    fmaxf((x), __int_as_float(__builtin_amdgcn_ds_swizzle(__float_as_int(x), (imm))))
#define SWZ_X1  0x041F
#define SWZ_X2  0x081F
#define SWZ_X4  0x101F
#define SWZ_X8  0x201F
#define SWZ_X16 0x401F

// f32 -> bf16 round-to-nearest-even (bit twiddle; host stays free of __bf16)
__device__ __forceinline__ unsigned short f32_to_bf16(float f) {
    unsigned int u = __float_as_uint(f);
    u += 0x7FFFu + ((u >> 16) & 1u);
    return (unsigned short)(u >> 16);
}

// ---------------------------------------------------------------------------
// 1) Convert both feature tensors to bf16 (one pass; all stays in 192MB L2).
// ---------------------------------------------------------------------------
__global__ void cvt_bf16_kernel(const float* __restrict__ fa32,
                                const float* __restrict__ fb32,
                                unsigned short* __restrict__ fa,
                                unsigned short* __restrict__ fb,
                                int nA, int nB) {
    int idx = blockIdx.x * blockDim.x + threadIdx.x;
    int stride = gridDim.x * blockDim.x;
    for (int i = idx; i < nA; i += stride) fa[i] = f32_to_bf16(fa32[i]);
    for (int i = idx; i < nB; i += stride) fb[i] = f32_to_bf16(fb32[i]);
}

// ---------------------------------------------------------------------------
// 2) Masks + masked softmax weights (one thread per batch row; tiny).
// ---------------------------------------------------------------------------
__global__ __launch_bounds__(128)
void prep_kernel(const int* __restrict__ tokens,          // [B,T]
                 const float* __restrict__ wTa,           // [B,T,1]
                 const int* __restrict__ vmask,           // [B,V]
                 const float* __restrict__ wVb,           // [B,V,1]
                 float* __restrict__ wA, float* __restrict__ wB,
                 float* __restrict__ mA, float* __restrict__ mB) {
    const int i = threadIdx.x;   // 0..127
    {   // text side: T=32
        float w[TT], msk[TT], mx = -__builtin_inff();
        for (int t = 0; t < TT; ++t) {
            float mm = (tokens[i * TT + t] != 0) ? 1.f : 0.f;
            msk[t] = mm;
            float ww = (mm > 0.f) ? wTa[i * TT + t] : -__builtin_inff();
            w[t] = ww;
            mx = fmaxf(mx, ww);
        }
        float s = 0.f;
        for (int t = 0; t < TT; ++t) s += __expf(w[t] - mx);
        float inv = 1.f / s;
        for (int t = 0; t < TT; ++t) {
            wA[i * TT + t] = __expf(w[t] - mx) * inv;
            mA[i * TT + t] = msk[t];
        }
    }
    {   // video side: V=64
        float w[VV], msk[VV], mx = -__builtin_inff();
        for (int v = 0; v < VV; ++v) {
            float mm = (vmask[i * VV + v] != 0) ? 1.f : 0.f;
            msk[v] = mm;
            float ww = (mm > 0.f) ? wVb[i * VV + v] : -__builtin_inff();
            w[v] = ww;
            mx = fmaxf(mx, ww);
        }
        float s = 0.f;
        for (int v = 0; v < VV; ++v) s += __expf(w[v] - mx);
        float inv = 1.f / s;
        for (int v = 0; v < VV; ++v) {
            wB[i * VV + v] = __expf(w[v] - mx) * inv;
            mB[i * VV + v] = msk[v];
        }
    }
}

// ---------------------------------------------------------------------------
// 3) Main kernel: one 128-thread block (4 waves) per (a,b) pair.
//    Wave w -> tTile = w>>1 (rows 16*tTile..+15), vPair = w&1 (cols 32*vPair..+31).
//    Two f32 accumulators per wave share one A fragment per K-step:
//    per k+=32: 2 A loads + 4 B loads -> 2 WMMAs.
// ---------------------------------------------------------------------------
__global__ __launch_bounds__(128)
void pair_score_kernel(const unsigned short* __restrict__ fa_us,  // [B,T,D] bf16
                       const unsigned short* __restrict__ fb_us,  // [B,V,D] bf16
                       const float* __restrict__ wA, const float* __restrict__ wB,
                       const float* __restrict__ mA, const float* __restrict__ mB,
                       const float* __restrict__ temp,
                       float* __restrict__ score) {
    __shared__ float wa_s[TT], ma_s[TT];
    __shared__ float wb_s[VV], mb_s[VV];
    __shared__ float a2b_part[TT][2];   // row-max per vPair
    __shared__ float b2a_part[VV][2];   // col-max per tTile
    __shared__ float red[128];

    const __bf16* __restrict__ fa = (const __bf16*)fa_us;
    const __bf16* __restrict__ fb = (const __bf16*)fb_us;

    const int a = blockIdx.x, b = blockIdx.y;
    const int tid = threadIdx.x;

    if (tid < TT) { wa_s[tid] = wA[a * TT + tid]; ma_s[tid] = mA[a * TT + tid]; }
    if (tid < VV) { wb_s[tid] = wB[b * VV + tid]; mb_s[tid] = mB[b * VV + tid]; }
    __syncthreads();

    const int wave  = tid >> 5;        // 0..3
    const int lane  = tid & 31;
    const int tTile = wave >> 1;       // 0..1
    const int vPair = wave & 1;        // 0..1 (32 columns each)
    const int m     = lane & 15;
    const int hi    = lane >> 4;

    // A-layout (16x32 bf16): lane holds row m, 16 contiguous K at offset hi*16
    const __bf16* Ap  = fa + ((size_t)a * TT + tTile * 16 + m) * DD + hi * 16;
    // B-layout (32x16 bf16, same k order as A): lane holds col m, 16 contiguous K at offset hi*16
    const __bf16* Bp0 = fb + ((size_t)b * VV + vPair * 32 + m) * DD + hi * 16;
    const __bf16* Bp1 = Bp0 + (size_t)16 * DD;   // columns +16

    v8f c0 = {}, c1 = {};
#pragma unroll 4
    for (int k = 0; k < DD; k += 32) {
        v16bf av = *(const v16bf*)(Ap + k);
        v16bf bv0 = *(const v16bf*)(Bp0 + k);
        v16bf bv1 = *(const v16bf*)(Bp1 + k);
        c0 = __builtin_amdgcn_wmma_f32_16x16x32_bf16(
                 false, av, false, bv0, (short)0, c0, false, false);
        c1 = __builtin_amdgcn_wmma_f32_16x16x32_bf16(
                 false, av, false, bv1, (short)0, c1, false, false);
    }

    // C element (r, lane): row t = tTile*16 + r + 8*hi; col N = m (+16 for c1)
    const float mb0 = mb_s[vPair * 32 + m];
    const float mb1 = mb_s[vPair * 32 + 16 + m];
    float cm0[8], cm1[8];
#pragma unroll
    for (int r = 0; r < 8; ++r) {
        float ma_r = ma_s[tTile * 16 + r + 8 * hi];
        cm0[r] = c0[r] * ma_r * mb0;
        cm1[r] = c1[r] * ma_r * mb1;
    }

    // Row max over this wave's 32 columns: in-lane pair max, then butterfly
    // across the 16-lane half (single ds_swizzle per step).
#pragma unroll
    for (int r = 0; r < 8; ++r) {
        float rm = fmaxf(cm0[r], cm1[r]);
        rm = SWZ_MAXF(rm, SWZ_X1);
        rm = SWZ_MAXF(rm, SWZ_X2);
        rm = SWZ_MAXF(rm, SWZ_X4);
        rm = SWZ_MAXF(rm, SWZ_X8);
        if (m == 0) a2b_part[tTile * 16 + r + 8 * hi][vPair] = rm;
    }
    // Col max: over r in-lane, then swap the two lane halves (M and M+8)
    float cx0 = cm0[0], cx1 = cm1[0];
#pragma unroll
    for (int r = 1; r < 8; ++r) {
        cx0 = fmaxf(cx0, cm0[r]);
        cx1 = fmaxf(cx1, cm1[r]);
    }
    cx0 = SWZ_MAXF(cx0, SWZ_X16);
    cx1 = SWZ_MAXF(cx1, SWZ_X16);
    if (hi == 0) {
        b2a_part[vPair * 32 + m][tTile]      = cx0;
        b2a_part[vPair * 32 + 16 + m][tTile] = cx1;
    }
    __syncthreads();

    // score[a,b] = 0.5 * (sum_t wA*max_v + sum_v wB*max_t) / temp
    float val = 0.f;
    if (tid < TT) {
        val = wa_s[tid] * fmaxf(a2b_part[tid][0], a2b_part[tid][1]);
    } else if (tid < TT + VV) {
        int v = tid - TT;
        val = wb_s[v] * fmaxf(b2a_part[v][0], b2a_part[v][1]);
    }
    red[tid] = val;
    __syncthreads();
    for (int s = 64; s > 0; s >>= 1) {
        if (tid < s) red[tid] += red[tid + s];
        __syncthreads();
    }
    if (tid == 0) score[a * NB + b] = red[0] * 0.5f / temp[0];
}

// ---------------------------------------------------------------------------
// 4) Final loss: row & column log-softmax diagonals, mean of the 256 losses.
// ---------------------------------------------------------------------------
__global__ __launch_bounds__(128)
void loss_kernel(const float* __restrict__ score, float* __restrict__ out) {
    __shared__ float red[128];
    const int i = threadIdx.x;
    float diag = score[i * NB + i];
    float rmax = -__builtin_inff(), cmax = -__builtin_inff();
    for (int j = 0; j < NB; ++j) {
        rmax = fmaxf(rmax, score[i * NB + j]);
        cmax = fmaxf(cmax, score[j * NB + i]);
    }
    float rs = 0.f, cs = 0.f;
    for (int j = 0; j < NB; ++j) {
        rs += __expf(score[i * NB + j] - rmax);
        cs += __expf(score[j * NB + i] - cmax);
    }
    float l = (__logf(rs) + rmax - diag) + (__logf(cs) + cmax - diag);
    red[i] = l;
    __syncthreads();
    for (int s = 64; s > 0; s >>= 1) {
        if (i < s) red[i] += red[i + s];
        __syncthreads();
    }
    if (i == 0) out[0] = red[0] * (1.f / 256.f);
}

// ---------------------------------------------------------------------------
// Launch
// ---------------------------------------------------------------------------
extern "C" void kernel_launch(void* const* d_in, const int* in_sizes, int n_in,
                              void* d_out, int out_size, void* d_ws, size_t ws_size,
                              hipStream_t stream) {
    (void)in_sizes; (void)n_in; (void)out_size; (void)ws_size;

    const int*   tokens = (const int*)d_in[0];    // text_tokens        [B,T] int
    const float* fa32   = (const float*)d_in[1];  // text features      [B,T,D]
    const float* wTa    = (const float*)d_in[2];  // text weights       [B,T,1]
    const float* fb32   = (const float*)d_in[3];  // video features     [B,V,D]
    const float* wVb    = (const float*)d_in[4];  // video weights      [B,V,1]
    const int*   vmask  = (const int*)d_in[5];    // video mask         [B,V] int
    const float* temp   = (const float*)d_in[6];  // temperature        [1]

    // Workspace layout (bytes)
    char* ws = (char*)d_ws;
    unsigned short* fa    = (unsigned short*)(ws + 0);          //  4,194,304 B (bf16 A)
    unsigned short* fb    = (unsigned short*)(ws + 4194304);    //  8,388,608 B (bf16 B)
    float*          score = (float*)(ws + 12582912);            //     65,536 B
    float*          wA    = (float*)(ws + 12648448);            //     16,384 B
    float*          wB    = (float*)(ws + 12664832);            //     32,768 B
    float*          mA    = (float*)(ws + 12697600);            //     16,384 B
    float*          mB    = (float*)(ws + 12713984);            //     32,768 B

    const int nA  = NB * TT * DD;  // 2,097,152
    const int nBf = NB * VV * DD;  // 4,194,304

    cvt_bf16_kernel<<<2048, 256, 0, stream>>>(fa32, fb32, fa, fb, nA, nBf);
    prep_kernel<<<1, 128, 0, stream>>>(tokens, wTa, vmask, wVb, wA, wB, mA, mB);
    pair_score_kernel<<<dim3(NB, NB), 128, 0, stream>>>(fa, fb, wA, wB, mA, mB, temp, score);
    loss_kernel<<<1, 128, 0, stream>>>(score, (float*)d_out);
}
